// Adaptive_MSELoss_MSE_80169859547709
// MI455X (gfx1250) — compile-verified, matrix operand order checked by reading, not run on hardware
//
#include <hip/hip_runtime.h>
#include <hip/hip_bf16.h>

// ---------------------------------------------------------------------------
// Adaptive MSE over shifts, MI455X (gfx1250).
// corr[s] computed with v_wmma_f32_16x16x32_bf16 using split-precision
// (hi/lo bf16) for ~fp32 accuracy; shift tile s = s0 + 16*m + n mapped to the
// 16x16 WMMA D tile via Toeplitz A/B fragments staged in LDS.
// ---------------------------------------------------------------------------

typedef __attribute__((ext_vector_type(16))) __bf16 v16bf;
typedef __attribute__((ext_vector_type(8)))  float  v8f;

#define NCH    1024        // B*C channels
#define L_IN   8192
#define L_TGT  10240
#define NSH    2049        // shifts
#define INV_N  (1.0f / 8388608.0f)

// reduction chunking: j0 in [0, 8448) step 32 -> 264 chunks, batches of 8
#define JB     8
#define NBATCH 33
#define UPAD   240         // front zero-pad of the u window (16*15)
#define UWIN   496         // UPAD + JB*32
#define VWIN   272         // (JB-1)*32 + 31 + 15 + slack

// workspace float offsets (total ~59 KB)
#define OFF_INPSQ 0
#define OFF_CORR  64
#define OFF_T2    2304
#define OFF_WIN   12608

union Frag { unsigned short h[16]; unsigned u[8]; v16bf bf; };

__device__ __forceinline__ void split_bf16(float x, unsigned short& hi,
                                           unsigned short& lo) {
    unsigned b = __float_as_uint(x);
    hi = (unsigned short)(b >> 16);                       // truncated bf16 hi
    float fh = __uint_as_float(((unsigned)hi) << 16);
    lo = (unsigned short)(__float_as_uint(x - fh) >> 16); // residual as bf16
}

// ---- zero the accumulated parts of the workspace (every call) --------------
__global__ void k_init(float* ws) {
    int i = blockIdx.x * blockDim.x + threadIdx.x;
    if (i == 0) ws[OFF_INPSQ] = 0.0f;
    if (i < NSH) ws[OFF_CORR + i] = 0.0f;
}

// ---- sum(input^2), exact fp32 ---------------------------------------------
__global__ void k_inpsq(const float* __restrict__ inp, float* ws) {
    __shared__ float red[256];
    float s = 0.0f;
    long long stride = (long long)gridDim.x * blockDim.x;
    for (long long i = (long long)blockIdx.x * blockDim.x + threadIdx.x;
         i < (long long)NCH * L_IN; i += stride) {
        float v = inp[i];
        s += v * v;
    }
    red[threadIdx.x] = s;
    __syncthreads();
    for (int off = 128; off > 0; off >>= 1) {
        if (threadIdx.x < off) red[threadIdx.x] += red[threadIdx.x + off];
        __syncthreads();
    }
    if (threadIdx.x == 0) atomicAdd(&ws[OFF_INPSQ], red[0]);
}

// ---- t2[l] = sum over channels of target^2 (coalesced in l) ---------------
__global__ void k_t2(const float* __restrict__ tgt, float* ws) {
    int l = blockIdx.x * blockDim.x + threadIdx.x;
    if (l >= L_TGT) return;
    float s = 0.0f;
    for (int c = 0; c < NCH; ++c) {
        float v = tgt[(long long)c * L_TGT + l];
        s += v * v;
    }
    ws[OFF_T2 + l] = s;
}

// ---- winsum[s] = sum_{t<L_IN} t2[s+t]; one block per shift (parallel) -----
__global__ void k_winsum(float* ws) {
    __shared__ float red[256];
    const int s = blockIdx.x;            // 0..NSH-1
    const float* t2 = ws + OFF_T2;
    float acc = 0.0f;
    for (int t = threadIdx.x; t < L_IN; t += 256) acc += t2[s + t];
    red[threadIdx.x] = acc;
    __syncthreads();
    for (int off = 128; off > 0; off >>= 1) {
        if (threadIdx.x < off) red[threadIdx.x] += red[threadIdx.x + off];
        __syncthreads();
    }
    if (threadIdx.x == 0) ws[OFF_WIN + s] = red[0];
}

// ---- main WMMA correlation kernel -----------------------------------------
// grid: (128 channel-groups, 9 shift-tiles), 256 threads = 8 waves,
// one channel per wave; each wave accumulates one 16x16 D tile (256 shifts).
__global__ void __launch_bounds__(256) k_corr(const float* __restrict__ inp,
                                              const float* __restrict__ tgt,
                                              float* ws) {
    __shared__ __align__(16) unsigned short sUH[8][512];
    __shared__ __align__(16) unsigned short sUL[8][512];
    __shared__ __align__(16) unsigned short sVH[8][288];
    __shared__ __align__(16) unsigned short sVL[8][288];

    const int tid    = threadIdx.x;
    const int wave   = tid >> 5;
    const int lane   = tid & 31;
    const int m      = lane & 15;   // A row (all lanes); D column n in the store
    const int laneHi = lane >> 4;

    const int c  = blockIdx.x * 8 + wave;   // channel
    const int s0 = blockIdx.y * 256;        // shift-tile base

    const float* uc = inp + (long long)c * L_IN;
    const float* vc = tgt + (long long)c * L_TGT;

    unsigned short* uh = sUH[wave];
    unsigned short* ul = sUL[wave];
    unsigned short* vh = sVH[wave];
    unsigned short* vl = sVL[wave];

    v8f acc = {};

    for (int batch = 0; batch < NBATCH; ++batch) {
        const int J0 = batch * (JB * 32);
        __syncthreads();
        // stage u window  t in [J0-240, J0+256), zero outside [0, L_IN)
        for (int i = lane; i < UWIN; i += 32) {
            int t = J0 - UPAD + i;
            float x = (t >= 0 && t < L_IN) ? uc[t] : 0.0f;
            unsigned short h, l;
            split_bf16(x, h, l);
            uh[i] = h; ul[i] = l;
        }
        // stage v window  idx in [J0+s0, J0+s0+272), zero past L_TGT
        for (int i = lane; i < VWIN; i += 32) {
            int idx = J0 + s0 + i;
            float x = (idx < L_TGT) ? vc[idx] : 0.0f;
            unsigned short h, l;
            split_bf16(x, h, l);
            vh[i] = h; vl[i] = l;
        }
        __syncthreads();

        for (int step = 0; step < JB; ++step) {
            const int j0rel = step * 32;
            Frag ah, alo, bh, blo;
#pragma unroll
            for (int p = 0; p < 8; ++p) {
                // per-lane K map from ISA 7.12.2 (16-bit A 16x32 layout):
                // pair p -> K = 2p + (p>=4 ? 8 : 0) + (lane>=16 ? 8 : 0)
                int kk = 2 * p + ((p & 4) ? 8 : 0) + (laneHi << 3);
                // A[m][k] = u[j0 + k - 16m]  (window starts at J0-UPAD)
                int la = j0rel + kk + UPAD - 16 * m;
                ah.u[p]  = *(const unsigned*)(uh + la);
                alo.u[p] = *(const unsigned*)(ul + la);
                // B[k][n] = v[j0 + s0 + k + n] (window starts at J0+s0)
                int lb = j0rel + kk + m;
                bh.h[2 * p]      = vh[lb];
                bh.h[2 * p + 1]  = vh[lb + 1];
                blo.h[2 * p]     = vl[lb];
                blo.h[2 * p + 1] = vl[lb + 1];
            }
            // split-precision: hi*hi + hi*lo + lo*hi  (lo*lo negligible)
            acc = __builtin_amdgcn_wmma_f32_16x16x32_bf16(
                false, ah.bf, false, bh.bf, (short)0, acc, false, false);
            acc = __builtin_amdgcn_wmma_f32_16x16x32_bf16(
                false, ah.bf, false, blo.bf, (short)0, acc, false, false);
            acc = __builtin_amdgcn_wmma_f32_16x16x32_bf16(
                false, alo.bf, false, bh.bf, (short)0, acc, false, false);
        }
    }

    // D layout (ISA 7.12.2): lanes 0-15: VGPR r -> M=r, N=lane;
    //                        lanes 16-31: M=r+8, N=lane-16.
    float* corr = ws + OFF_CORR;
#pragma unroll
    for (int r = 0; r < 8; ++r) {
        int s = s0 + 16 * (r + 8 * laneHi) + m;
        if (s < NSH) atomicAdd(&corr[s], acc[r]);
    }
}

// ---- losses + first-occurrence argmin -------------------------------------
__global__ void k_final(const float* __restrict__ ws, float* out) {
    __shared__ float bv[256];
    __shared__ int   bi[256];
    const float  isq  = ws[OFF_INPSQ];
    const float* corr = ws + OFF_CORR;
    const float* win  = ws + OFF_WIN;
    float best = 3.4e38f;
    int   besti = NSH;
    for (int s = threadIdx.x; s < NSH; s += 256) {
        float loss = (isq - 2.0f * corr[s] + win[s]) * INV_N;
        if (loss < best) { best = loss; besti = s; }   // indices increase
    }
    bv[threadIdx.x] = best;
    bi[threadIdx.x] = besti;
    __syncthreads();
    for (int off = 128; off > 0; off >>= 1) {
        if (threadIdx.x < off) {
            float ov = bv[threadIdx.x + off];
            int   oi = bi[threadIdx.x + off];
            if (ov < bv[threadIdx.x] ||
                (ov == bv[threadIdx.x] && oi < bi[threadIdx.x])) {
                bv[threadIdx.x] = ov;
                bi[threadIdx.x] = oi;
            }
        }
        __syncthreads();
    }
    if (threadIdx.x == 0) {
        out[0] = bv[0];
        out[1] = (float)bi[0];
    }
}

extern "C" void kernel_launch(void* const* d_in, const int* in_sizes, int n_in,
                              void* d_out, int out_size, void* d_ws,
                              size_t ws_size, hipStream_t stream) {
    (void)in_sizes; (void)n_in; (void)out_size; (void)ws_size;
    const float* inp = (const float*)d_in[0];   // (16,64,8192)  fp32
    const float* tgt = (const float*)d_in[1];   // (16,64,10240) fp32
    float* ws  = (float*)d_ws;                  // needs ~60 KB
    float* out = (float*)d_out;                 // [min_loss, min_index]

    k_init  <<<9,    256, 0, stream>>>(ws);
    k_inpsq <<<1024, 256, 0, stream>>>(inp, ws);
    k_t2    <<<(L_TGT + 255) / 256, 256, 0, stream>>>(tgt, ws);
    k_winsum<<<NSH, 256, 0, stream>>>(ws);
    dim3 grid(128, 9);
    k_corr  <<<grid, 256, 0, stream>>>(inp, tgt, ws);
    k_final <<<1,    256, 0, stream>>>(ws, out);
}
